// SimpleLSTM_79834852098246
// MI455X (gfx1250) — matrix operand order repaired and checked by reading, not compiled, not run on hardware
//
#include <hip/hip_runtime.h>
#include <hip/hip_bf16.h>

// ---------------------------------------------------------------------------
// LSTM with projection on MI455X (gfx1250, wave32, WMMA)
//   B=32, L=512, E=1024, H=1024, P=512
// Phase 1: xg = x @ W_ih^T + b_ih + b_hh   (bf16 WMMA, f32 accumulate)
// Phase 2: persistent 16-WG scan kernel; each WG owns 64 hidden columns,
//          W_hh slice staged into LDS via Tensor Data Mover, per-step
//          cross-WGP reduction of h through L2 atomics + cluster barrier.
// ---------------------------------------------------------------------------

typedef __attribute__((ext_vector_type(16))) __bf16 v16bf;
typedef __attribute__((ext_vector_type(8)))  __bf16 bf16x8;
typedef __attribute__((ext_vector_type(8)))  float  v8f;
typedef __attribute__((ext_vector_type(4)))  unsigned int u32x4;
typedef __attribute__((ext_vector_type(8)))  int    i32x8;
typedef __attribute__((ext_vector_type(4)))  int    i32x4;

#define NWG 16   // one cluster of 16 workgroups (one WGP each)

static __device__ inline v16bf cat8(bf16x8 lo, bf16x8 hi) {
  return __builtin_shufflevector(lo, hi, 0,1,2,3,4,5,6,7,8,9,10,11,12,13,14,15);
}

static __device__ inline v8f wmma_bf16(v16bf a, v16bf b, v8f c) {
  return __builtin_amdgcn_wmma_f32_16x16x32_bf16(false, a, false, b, (short)0, c,
                                                 false, false);
}

static __device__ inline float fsig(float x) {
  return __builtin_amdgcn_rcpf(1.0f + __expf(-x));
}
static __device__ inline float ftanh(float x) {
  float e = __expf(-2.0f * x);
  return (1.0f - e) * __builtin_amdgcn_rcpf(1.0f + e);
}

// ---- Tensor Data Mover: 2-D bf16 tile (tile1 rows x tile0 elems) -> LDS ----
static __device__ inline void tdm_load_2d(unsigned lds_addr, unsigned long long ga,
                                          unsigned dim0, unsigned dim1,
                                          unsigned long long stride0,
                                          unsigned tile0, unsigned tile1) {
#if __has_builtin(__builtin_amdgcn_tensor_load_to_lds)
  u32x4 g0;
  g0[0] = 1u;                                            // count=1 (valid D#)
  g0[1] = lds_addr;                                      // LDS byte address
  g0[2] = (unsigned)(ga & 0xffffffffull);                // global_addr lo
  g0[3] = (unsigned)((ga >> 32) & 0x1ffffffull)          // global_addr hi
        | (2u << 30);                                    // type = 2 ("image")
  i32x8 g1;
  g1[0] = (int)(1u << 16);                               // data_size = 2 bytes
  g1[1] = (int)((dim0 & 0xffffu) << 16);                 // tensor_dim0 [15:0]
  g1[2] = (int)((dim0 >> 16) | ((dim1 & 0xffffu) << 16));// dim0 hi | dim1 lo
  g1[3] = (int)((dim1 >> 16) | (tile0 << 16));           // dim1 hi | tile_dim0
  g1[4] = (int)(tile1 & 0xffffu);                        // tile_dim1 (tile2=0)
  g1[5] = (int)(stride0 & 0xffffffffull);                // dim0_stride lo
  g1[6] = (int)((stride0 >> 32) & 0xffffu);              // dim0_stride hi
  g1[7] = 0;
  i32x4 z4 = {0, 0, 0, 0};                               // groups 2/3 unused (2-D)
  i32x8 z8 = {0, 0, 0, 0, 0, 0, 0, 0};
  // 6-arg form (this toolchain): (g0, g1, g2, g3, g4, cpol)
  __builtin_amdgcn_tensor_load_to_lds(g0, g1, z4, z4, z8, 0);
#else
  (void)lds_addr; (void)ga; (void)dim0; (void)dim1;
  (void)stride0; (void)tile0; (void)tile1;
#endif
}

// ---- device-wide sense-reversing barrier + HW cluster barrier fast path ----
static __device__ inline void grid_barrier(unsigned* cnt, unsigned* gen) {
  __threadfence();                 // flush this thread's stores device-wide
  __syncthreads();
  if (threadIdx.x == 0) {
    // CDNA5 cluster barrier (s_barrier_signal/-wait -3). NOP when the grid is
    // not dispatched as a workgroup cluster; real HW sync when it is.
    __builtin_amdgcn_s_cluster_barrier();
    unsigned g   = __hip_atomic_load(gen, __ATOMIC_RELAXED, __HIP_MEMORY_SCOPE_AGENT);
    unsigned old = __hip_atomic_fetch_add(cnt, 1u, __ATOMIC_ACQ_REL, __HIP_MEMORY_SCOPE_AGENT);
    if (old == NWG - 1u) {
      __hip_atomic_store(cnt, 0u, __ATOMIC_RELAXED, __HIP_MEMORY_SCOPE_AGENT);
      __hip_atomic_fetch_add(gen, 1u, __ATOMIC_RELEASE, __HIP_MEMORY_SCOPE_AGENT);
    } else {
      while (__hip_atomic_load(gen, __ATOMIC_ACQUIRE, __HIP_MEMORY_SCOPE_AGENT) == g)
        __builtin_amdgcn_s_sleep(1);
    }
  }
  __syncthreads();
}

// ---------------------------------------------------------------------------
// f32 -> bf16 conversion
// ---------------------------------------------------------------------------
__global__ void cvt_bf16_kernel(const float* __restrict__ s, __bf16* __restrict__ d,
                                long n) {
  long i = (long)blockIdx.x * blockDim.x + threadIdx.x;
  long stride = (long)gridDim.x * blockDim.x;
  for (; i < n; i += stride) d[i] = (__bf16)s[i];
}

// ---------------------------------------------------------------------------
// Phase 1: xg[m=b*512+t][4096] = X[m][:] @ W_ih^T + (b_ih + b_hh)
//   block = 256 threads (8 waves); each wave -> one 16x16 output tile, K=1024
// ---------------------------------------------------------------------------
__global__ void xg_gemm_kernel(const __bf16* __restrict__ Xb,
                               const __bf16* __restrict__ Wb,
                               const float* __restrict__ b_ih,
                               const float* __restrict__ b_hh,
                               float* __restrict__ xg) {
  const int w    = threadIdx.x >> 5;
  const int lane = threadIdx.x & 31;
  const int sel  = lane >> 4;          // K-half select per CDNA5 A/B layouts
  const int ln   = lane & 15;
  const long tile = (long)blockIdx.x * 8 + w;   // 1024 Mtiles x 256 Ntiles
  const int mt = (int)(tile >> 8);
  const int nt = (int)(tile & 255);

  const int col = nt * 16 + ln;
  const float bias = b_ih[col] + b_hh[col];
  v8f acc;
#pragma unroll
  for (int r = 0; r < 8; ++r) acc[r] = bias;

  const __bf16* arow = Xb + (size_t)(mt * 16 + ln) * 1024;
  const __bf16* brow = Wb + (size_t)col * 1024;

#pragma unroll 4
  for (int kk = 0; kk < 32; ++kk) {
    const int k = kk * 32;
    // A: 16x32 bf16 tile of X (lane<16: K{0-7,16-23}; lane>=16: K{8-15,24-31})
    v16bf A = cat8(*(const bf16x8*)(arow + k + sel * 8),
                   *(const bf16x8*)(arow + k + 16 + sel * 8));
    // B: 32x16 tile of W_ih^T (lane<16: K{0-7,16-23}; lane>=16: K{8-15,24-31})
    v16bf B = cat8(*(const bf16x8*)(brow + k + sel * 8),
                   *(const bf16x8*)(brow + k + 16 + sel * 8));
    acc = wmma_bf16(A, B, acc);
  }

#pragma unroll
  for (int rr = 0; rr < 8; ++rr) {
    const int mrow = mt * 16 + rr + 8 * sel;   // D layout: M = rr + 8*(lane>=16)
    xg[(size_t)mrow * 4096 + nt * 16 + ln] = acc[rr];
  }
}

// ---------------------------------------------------------------------------
// zero ACC double-buffer + barrier counters
// ---------------------------------------------------------------------------
__global__ void zero_ws_kernel(float* __restrict__ acc, unsigned* __restrict__ sync) {
  int i = blockIdx.x * blockDim.x + threadIdx.x;
  if (i < 32768) acc[i] = 0.0f;
  if (i < 8) sync[i] = 0u;
}

// ---------------------------------------------------------------------------
// Phase 2: persistent recurrent scan. 16 WGs x 256 threads (8 waves/WG).
//   WG wg owns hidden columns [wg*64, wg*64+64).
//   wave w: m = w&1 (batch half), cg = w>>1 (16-column group)
//     -> 4 gate accumulators (i,f,g,o) + persistent cell state in VGPRs.
//   LDS: W_hh slice 256KB (TDM) | h_prev bf16 32KB | a=o*tanh(c) bf16 4KB
// ---------------------------------------------------------------------------
__global__ void lstm_scan_kernel(const float* __restrict__ xg,
                                 const __bf16* __restrict__ whh,
                                 const __bf16* __restrict__ whr,
                                 float* __restrict__ accbuf,
                                 unsigned* __restrict__ sync,
                                 float* __restrict__ out) {
  extern __shared__ char smem[];
  __bf16* whh_lds = (__bf16*)smem;                 // [4 gates][64 rows][512 K]
  __bf16* h_lds   = (__bf16*)(smem + 262144);      // [32 b][512 k]
  __bf16* a_buf   = (__bf16*)(smem + 294912);      // [32 b][64 k]

  const int tid  = threadIdx.x;
  const int wg   = blockIdx.x;          // 0..15 (workgroup-in-cluster)
  const int w    = tid >> 5;
  const int lane = tid & 31;
  const int sel  = lane >> 4;
  const int ln   = lane & 15;
  const int m    = w & 1;
  const int cg   = w >> 1;

  float* ACC[2] = { accbuf, accbuf + 16384 };
  unsigned* cnt = sync;
  unsigned* gen = sync + 1;

  // -- one-time: stage this WG's W_hh slice into LDS with the TDM -----------
  if (w == 0) {
    const unsigned lb = (unsigned)(uintptr_t)whh_lds;
#pragma unroll
    for (int gi = 0; gi < 4; ++gi) {
      unsigned long long ga =
          (unsigned long long)(uintptr_t)(whh + ((size_t)gi * 1024 + wg * 64) * 512);
      // 64 rows x 512 bf16, row stride 512 elements -> 64KB per gate block
      tdm_load_2d(lb + (unsigned)gi * 65536u, ga, 512u, 4096u, 512ull, 512u, 64u);
    }
#if __has_builtin(__builtin_amdgcn_tensor_load_to_lds)
    __builtin_amdgcn_s_wait_tensorcnt(0);
#endif
  }
#if !__has_builtin(__builtin_amdgcn_tensor_load_to_lds)
  {  // fallback: cooperative copy (4 contiguous 64KB blocks)
    for (int gi = 0; gi < 4; ++gi) {
      const int4* s = (const int4*)(whh + ((size_t)gi * 1024 + wg * 64) * 512);
      int4* d = (int4*)(whh_lds + (size_t)gi * 32768);
      for (int i = tid; i < 4096; i += 256) d[i] = s[i];
    }
  }
#endif
  __syncthreads();

  // persistent cell state: c[b = m*16 + rr + 8*sel][col = wg*64 + cg*16 + ln]
  v8f cfrag = {};

  for (int t = 0; t < 512; ++t) {
    const int cur = t & 1, nxt = cur ^ 1;
    const float* hsrc = ACC[cur];

    // ---- Phase A: stage h_prev (f32 -> bf16 LDS) + emit out[:, t-1, :] ----
    {
      const float4* s4 = (const float4*)hsrc;
#pragma unroll
      for (int rep = 0; rep < 16; ++rep) {
        const int idx = rep * 256 + tid;         // 4096 float4
        float4 v = s4[idx];
        const int b0 = idx * 4;
        h_lds[b0 + 0] = (__bf16)v.x; h_lds[b0 + 1] = (__bf16)v.y;
        h_lds[b0 + 2] = (__bf16)v.z; h_lds[b0 + 3] = (__bf16)v.w;
      }
      if (t > 0) {
#pragma unroll
        for (int rep = 0; rep < 4; ++rep) {
          const int idx = rep * 256 + tid;       // 32 b x 32 p slice
          const int b = idx >> 5, p = wg * 32 + (idx & 31);
          out[((size_t)b * 512 + (t - 1)) * 512 + p] = hsrc[b * 512 + p];
        }
      }
    }
    grid_barrier(cnt, gen);   // every WG finished reading ACC[cur]

    // ---- Phase C: zero my 1/16 of ACC[cur] (accumulator for step t+1) ----
    ((float4*)(ACC[cur] + wg * 1024))[tid] = make_float4(0.f, 0.f, 0.f, 0.f);

    // ---- Phase D: gates = xg[b][t][:] + h_prev @ W_hh^T ------------------
    v8f ga_i, ga_f, ga_g, ga_o;
#pragma unroll
    for (int rr = 0; rr < 8; ++rr) {
      const int b = m * 16 + rr + 8 * sel;
      const float* xr = xg + ((size_t)b * 512 + t) * 4096 + (wg * 64 + cg * 16 + ln);
      ga_i[rr] = xr[0];     ga_f[rr] = xr[1024];
      ga_g[rr] = xr[2048];  ga_o[rr] = xr[3072];
    }
#pragma unroll 4
    for (int kk = 0; kk < 16; ++kk) {
      const int K0 = kk * 32;
      const __bf16* hr = h_lds + (size_t)(m * 16 + ln) * 512 + K0;
      v16bf A = cat8(*(const bf16x8*)(hr + sel * 8),
                     *(const bf16x8*)(hr + 16 + sel * 8));
      const __bf16* br = whh_lds + (size_t)(cg * 16 + ln) * 512 + K0 + sel * 8;
      v16bf B0 = cat8(*(const bf16x8*)(br),             *(const bf16x8*)(br + 16));
      v16bf B1 = cat8(*(const bf16x8*)(br + 32768),     *(const bf16x8*)(br + 32768 + 16));
      v16bf B2 = cat8(*(const bf16x8*)(br + 65536),     *(const bf16x8*)(br + 65536 + 16));
      v16bf B3 = cat8(*(const bf16x8*)(br + 98304),     *(const bf16x8*)(br + 98304 + 16));
      ga_i = wmma_bf16(A, B0, ga_i);
      ga_f = wmma_bf16(A, B1, ga_f);
      ga_g = wmma_bf16(A, B2, ga_g);
      ga_o = wmma_bf16(A, B3, ga_o);
    }
    // elementwise LSTM cell (all four gates element-aligned in this lane)
#pragma unroll
    for (int rr = 0; rr < 8; ++rr) {
      const float iv = fsig(ga_i[rr]);
      const float fv = fsig(ga_f[rr]);
      const float gv = ftanh(ga_g[rr]);
      const float ov = fsig(ga_o[rr]);
      const float c  = fv * cfrag[rr] + iv * gv;
      cfrag[rr] = c;
      const float a = ov * ftanh(c);
      const int b = m * 16 + rr + 8 * sel;
      a_buf[b * 64 + cg * 16 + ln] = (__bf16)a;
    }
    __syncthreads();

    // ---- Phase E: projection partial  h += a @ W_hr^T[ , kslice] ---------
    {
      const int q = w >> 1;     // 4 waves per batch-half, 8 N-tiles each
#pragma unroll
      for (int ti = 0; ti < 8; ++ti) {
        const int nt = q * 8 + ti;
        v8f pa = {};
#pragma unroll
        for (int ks = 0; ks < 2; ++ks) {
          const int K0 = ks * 32;
          const __bf16* ar = a_buf + (size_t)(m * 16 + ln) * 64 + K0;
          v16bf A = cat8(*(const bf16x8*)(ar + sel * 8),
                         *(const bf16x8*)(ar + 16 + sel * 8));
          const __bf16* br = whr + (size_t)(nt * 16 + ln) * 1024
                                 + wg * 64 + K0 + sel * 8;
          v16bf B = cat8(*(const bf16x8*)br, *(const bf16x8*)(br + 16));
          pa = wmma_bf16(A, B, pa);
        }
        float* dst = ACC[nxt];
#pragma unroll
        for (int rr = 0; rr < 8; ++rr) {
          const int b = m * 16 + rr + 8 * sel;
          const int p = nt * 16 + ln;
          __hip_atomic_fetch_add(&dst[b * 512 + p], pa[rr],
                                 __ATOMIC_RELAXED, __HIP_MEMORY_SCOPE_AGENT);
        }
      }
      if (t + 1 < 512)  // warm L2/WGP$ for next step's xg slice
        __builtin_prefetch(xg + ((size_t)(m * 16 + ln) * 512 + (t + 1)) * 4096
                              + wg * 64, 0, 1);
    }
    grid_barrier(cnt, gen);   // h(t) fully reduced in ACC[nxt]
  }

  // ---- final timestep output (h(511) lives in ACC[0]) --------------------
  {
    const float* hf = ACC[0];
#pragma unroll
    for (int rep = 0; rep < 4; ++rep) {
      const int idx = rep * 256 + tid;
      const int b = idx >> 5, p = wg * 32 + (idx & 31);
      out[((size_t)b * 512 + 511) * 512 + p] = hf[b * 512 + p];
    }
  }
}

// ---------------------------------------------------------------------------
// host-side launcher
// ---------------------------------------------------------------------------
extern "C" void kernel_launch(void* const* d_in, const int* in_sizes, int n_in,
                              void* d_out, int out_size, void* d_ws, size_t ws_size,
                              hipStream_t stream) {
  (void)in_sizes; (void)n_in; (void)out_size; (void)ws_size;
  const float* x_emb = (const float*)d_in[0];   // [32,512,1024]
  const float* W_ih  = (const float*)d_in[1];   // [4096,1024]
  const float* W_hh  = (const float*)d_in[2];   // [4096,512]
  const float* W_hr  = (const float*)d_in[3];   // [512,1024]
  const float* b_ih  = (const float*)d_in[4];   // [4096]
  const float* b_hh  = (const float*)d_in[5];   // [4096]
  float* out = (float*)d_out;                   // [32,512,512]

  char* ws = (char*)d_ws;
  size_t off = 0;
  float*  xg     = (float*)(ws + off);  off += (size_t)16384 * 4096 * 4; // 256MB
  __bf16* x_bf   = (__bf16*)(ws + off); off += (size_t)16384 * 1024 * 2; //  32MB
  __bf16* wih_bf = (__bf16*)(ws + off); off += (size_t)4096 * 1024 * 2;  //   8MB
  __bf16* whh_bf = (__bf16*)(ws + off); off += (size_t)4096 * 512 * 2;   //   4MB
  __bf16* whr_bf = (__bf16*)(ws + off); off += (size_t)512 * 1024 * 2;   //   1MB
  float*  accbuf = (float*)(ws + off);  off += (size_t)2 * 16384 * 4;    // 128KB
  unsigned* sync = (unsigned*)(ws + off); off += 256;

  // f32 -> bf16 conversions
  cvt_bf16_kernel<<<4096, 256, 0, stream>>>(x_emb, x_bf,  (long)16384 * 1024);
  cvt_bf16_kernel<<<2048, 256, 0, stream>>>(W_ih,  wih_bf, (long)4096 * 1024);
  cvt_bf16_kernel<<<1024, 256, 0, stream>>>(W_hh,  whh_bf, (long)4096 * 512);
  cvt_bf16_kernel<<<512,  256, 0, stream>>>(W_hr,  whr_bf, (long)512 * 1024);

  // Phase 1: xg GEMM  (1024 x 256 tiles, 8 tiles per 256-thread block)
  xg_gemm_kernel<<<32768, 256, 0, stream>>>(x_bf, wih_bf, b_ih, b_hh, xg);

  // zero h double-buffer + barrier state
  zero_ws_kernel<<<128, 256, 0, stream>>>(accbuf, sync);

  // Phase 2: persistent recurrent scan — 16 WGs (ideally one 16-WG cluster),
  // 292KB dynamic LDS per WGP.
  lstm_scan_kernel<<<dim3(NWG), dim3(256), 299008, stream>>>(
      xg, whh_bf, whr_bf, accbuf, sync, out);
}
